// CLVPAttention_70420283785345
// MI455X (gfx1250) — matrix operand order repaired and checked by reading, not compiled, not run on hardware
//
#include <hip/hip_runtime.h>
#include <hip/hip_bf16.h>

// ---------------------------------------------------------------------------
// CLVP attention block for MI455X (gfx1250), bf16 WMMA path + async-LDS DMA.
//   k0: fp32 -> bf16 conversion of hidden states / weights (one-time, ~2us)
//   k1: fused QKV projection + scale + partial rotary -> bf16 [B,H,S,D]
//   k2: flash attention (QK^T, online softmax, PV) -> bf16 ctx [B,S,E]
//   k3: output projection + bias -> f32 out [B,S,E]
// Matrix math: v_wmma_f32_16x16x32_bf16.  Tile staging: GLOBAL_LOAD_ASYNC_TO_LDS.
// ---------------------------------------------------------------------------

constexpr int CB = 4, CS = 2048, CE = 1024, CH = 16, CD = 64, CROT = 32;
constexpr int CBS = CB * CS;           // 8192 token rows

typedef __bf16 bf16_t;
typedef bf16_t v16bf __attribute__((ext_vector_type(16)));
typedef float  v8f   __attribute__((ext_vector_type(8)));
typedef int    v4i   __attribute__((ext_vector_type(4)));

#if defined(__gfx1250__) && __has_builtin(__builtin_amdgcn_global_load_async_to_lds_b128)
#define ASYNC_LDS 1
#endif

// 16-byte global -> LDS copy. Async DMA (ASYNCcnt) when available.
// Builtin signature (from clang diagnostic): (v4i AS1* src, v4i AS3* dst, Ii, Ii).
__device__ inline void cp_async16(bf16_t* lds_dst, const bf16_t* gsrc) {
#ifdef ASYNC_LDS
  typedef __attribute__((address_space(1))) v4i* gp_t;
  typedef __attribute__((address_space(3))) v4i* lp_t;
  __builtin_amdgcn_global_load_async_to_lds_b128((gp_t)gsrc, (lp_t)lds_dst, 0, 0);
#else
  *(v4i*)lds_dst = *(const v4i*)gsrc;
#endif
}

__device__ inline void wait_async() {
#ifdef ASYNC_LDS
#if __has_builtin(__builtin_amdgcn_s_wait_asynccnt)
  __builtin_amdgcn_s_wait_asynccnt(0);
#else
  asm volatile("s_wait_asynccnt 0" ::: "memory");
#endif
#endif
}

__device__ inline bf16_t f2bf(float f) { return (bf16_t)f; }

__device__ inline v8f wmma_bf16(v16bf a, v16bf b, v8f c) {
  return __builtin_amdgcn_wmma_f32_16x16x32_bf16(
      /*neg_a=*/false, a, /*neg_b=*/false, b,
      /*c_mod=*/(short)0, c, /*reuse_a=*/false, /*reuse_b=*/false);
}

// A fragment (16x32 bf16, M x K). rowbase = tile + m*ldk + kstep, m = base + (lane&15).
// lane<16: elems 0..15 -> K 0..15; lane>=16: elems 0..15 -> K 16..31.
// Both 8-element runs are 16B-aligned and contiguous -> ds_load_b128 pairs.
__device__ inline v16bf ldsA(const bf16_t* rowbase, int lane) {
  const int koff = (lane >> 4) << 4;
  v16bf a;
#pragma unroll
  for (int h = 0; h < 2; ++h)
#pragma unroll
    for (int e = 0; e < 8; ++e)
      a[h * 8 + e] = rowbase[h * 8 + koff + e];
  return a;
}

// B fragment (32x16 bf16, K x N) from a tile stored [n][k] (k contiguous).
// B[k][n] = tile[n][k]; lane<16 -> K 0..15, lane>=16 -> K 16..31; n = lane&15.
__device__ inline v16bf ldsB_nk(const bf16_t* tile, int n0, int k0, int lane, int ldk) {
  const bf16_t* p = tile + (size_t)(n0 + (lane & 15)) * ldk + k0 + ((lane >> 4) << 4);
  v16bf b;
#pragma unroll
  for (int i = 0; i < 16; ++i) b[i] = p[i];
  return b;
}

// B fragment from a tile stored [k][n] (n contiguous): strided in k.
__device__ inline v16bf ldsB_kn(const bf16_t* tile, int n0, int k0, int lane, int ldn) {
  const bf16_t* p = tile + (size_t)(k0 + ((lane >> 4) << 4)) * ldn + n0 + (lane & 15);
  v16bf b;
#pragma unroll
  for (int i = 0; i < 16; ++i) b[i] = p[(size_t)i * ldn];
  return b;
}

// ---------------------------------------------------------------------------
// Kernel 0: fp32 -> bf16 conversion.  n must be a multiple of 2048.
// ---------------------------------------------------------------------------
__global__ void __launch_bounds__(256) convert_bf16_kernel(
    const float* __restrict__ src, bf16_t* __restrict__ dst, int n) {
  const int i = (blockIdx.x * 256 + threadIdx.x) * 8;
  if (i >= n) return;
#pragma unroll
  for (int j = 0; j < 8; ++j) dst[i + j] = f2bf(src[i + j]);
}

// ---------------------------------------------------------------------------
// Kernel 1: QKV projection + rotary.  grid = (CBS/128, CH, 3), block = 256.
// C[m,n] = sum_k X[m,k] * W[n,k];  N-tile (64) == one head, so rotary pairs
// (d, d+16) sit in adjacent accumulators of the same lane.
// ---------------------------------------------------------------------------
__global__ void __launch_bounds__(256) qkv_rope_kernel(
    const bf16_t* __restrict__ X, const float* __restrict__ rope,
    const bf16_t* __restrict__ Wq, const bf16_t* __restrict__ Wk,
    const bf16_t* __restrict__ Wv,
    bf16_t* __restrict__ Qb, bf16_t* __restrict__ Kb, bf16_t* __restrict__ Vb) {
  constexpr int LDK = 40;      // 80 B rows: 16B-aligned, 16 distinct banks
  __shared__ bf16_t Xs[128 * LDK];
  __shared__ bf16_t Ws[64 * LDK];

  const int tid = threadIdx.x;
  const int wid = tid >> 5, lane = tid & 31;
  const int m0 = blockIdx.x * 128;
  const int head = blockIdx.y;
  const int n0 = head * 64;
  const int which = blockIdx.z;
  const bf16_t* W = (which == 0) ? Wq : (which == 1) ? Wk : Wv;
  bf16_t* dst = (which == 0) ? Qb : (which == 1) ? Kb : Vb;
  const float alpha = (which == 0) ? 0.125f : 1.0f;   // D^-0.5 on q

  v8f acc[4] = {};

  for (int k0 = 0; k0 < CE; k0 += 32) {
    __syncthreads();
    // X tile: 128 rows x 32 k = 512 x b128 chunks (2 per thread).
#pragma unroll
    for (int t = 0; t < 2; ++t) {
      const int ci = tid + t * 256;
      const int r = ci >> 2, ch = ci & 3;
      cp_async16(&Xs[r * LDK + ch * 8], &X[(size_t)(m0 + r) * CE + k0 + ch * 8]);
    }
    // W tile: 64 rows x 32 k = 256 x b128 chunks (1 per thread).
    {
      const int r = tid >> 2, ch = tid & 3;
      cp_async16(&Ws[r * LDK + ch * 8], &W[(size_t)(n0 + r) * CE + k0 + ch * 8]);
    }
    wait_async();
    __syncthreads();
    v16bf a = ldsA(&Xs[(wid * 16 + (lane & 15)) * LDK], lane);
#pragma unroll
    for (int j = 0; j < 4; ++j) {
      v16bf b = ldsB_nk(Ws, j * 16, 0, lane, LDK);
      acc[j] = wmma_bf16(a, b, acc[j]);
    }
  }

  // Epilogue: scale, rotary on dims [0,32), store bf16 [B,H,S,D].
  const int c = lane & 15;
#pragma unroll
  for (int r = 0; r < 8; ++r) {
    const int m_local = ((lane >> 4) << 3) + r;
    const int gm = m0 + wid * 16 + m_local;     // global token row in [0, B*S)
    const int bb = gm >> 11;                    // / CS
    const int s  = gm & (CS - 1);               // % CS
    float t0 = acc[0][r] * alpha;
    float t1 = acc[1][r] * alpha;
    float t2 = acc[2][r] * alpha;
    float t3 = acc[3][r] * alpha;
    const float f0 = rope[s * CROT + c];
    const float f1 = rope[s * CROT + 16 + c];
    const float o0 = t0 * __cosf(f0) - t1 * __sinf(f0);   // d = c
    const float o1 = t1 * __cosf(f1) + t0 * __sinf(f1);   // d = c + 16
    const size_t base = (((size_t)bb * CH + head) * CS + s) * CD;
    dst[base +  0 + c] = f2bf(o0);
    dst[base + 16 + c] = f2bf(o1);
    dst[base + 32 + c] = f2bf(t2);
    dst[base + 48 + c] = f2bf(t3);
  }
}

// ---------------------------------------------------------------------------
// Kernel 2: flash attention.  grid = (CS/64, CH, CB), block = 128 (4 waves).
// Each wave owns a 16-query strip; stream keys/values in 64-row tiles via
// async LDS DMA.
// ---------------------------------------------------------------------------
__global__ void __launch_bounds__(128) attention_kernel(
    const bf16_t* __restrict__ Qb, const bf16_t* __restrict__ Kb,
    const bf16_t* __restrict__ Vb, bf16_t* __restrict__ ctx) {
  constexpr int LQ = 72;       // 144 B rows: 16B-aligned, distinct banks
  __shared__ bf16_t Qs[64 * LQ];
  __shared__ bf16_t Ks[64 * LQ];
  __shared__ bf16_t Vs[64 * LQ];
  __shared__ bf16_t Ps[4 * 16 * LQ];            // per-wave P tile (16 x 64)

  const int tid = threadIdx.x;
  const int wid = tid >> 5, lane = tid & 31;
  const int q0 = blockIdx.x * 64;
  const int head = blockIdx.y, b = blockIdx.z;
  const size_t hb = ((size_t)b * CH + head) * CS * CD;
  const bf16_t* Qg = Qb + hb + (size_t)q0 * CD;
  const bf16_t* Kg = Kb + hb;
  const bf16_t* Vg = Vb + hb;

  // Q tile: 64 rows x 64 d = 512 x b128 chunks (4 per thread).
#pragma unroll
  for (int t = 0; t < 4; ++t) {
    const int ci = tid + t * 128;
    const int r = ci >> 3, ch = ci & 7;
    cp_async16(&Qs[r * LQ + ch * 8], &Qg[(size_t)r * CD + ch * 8]);
  }
  wait_async();

  v8f o[4] = {};
  float mrun[8], lrun[8];
#pragma unroll
  for (int r = 0; r < 8; ++r) { mrun[r] = -1e30f; lrun[r] = 0.0f; }
  bf16_t* Pw = &Ps[wid * 16 * LQ];
  const int qrow_base = wid * 16;
  const int c = lane & 15;

  for (int kb0 = 0; kb0 < CS; kb0 += 64) {
    __syncthreads();                            // prev iter's reads complete
#pragma unroll
    for (int t = 0; t < 4; ++t) {
      const int ci = tid + t * 128;
      const int r = ci >> 3, ch = ci & 7;
      cp_async16(&Ks[r * LQ + ch * 8], &Kg[(size_t)(kb0 + r) * CD + ch * 8]);
      cp_async16(&Vs[r * LQ + ch * 8], &Vg[(size_t)(kb0 + r) * CD + ch * 8]);
    }
    wait_async();
    __syncthreads();

    // S = Q K^T (16q x 64k per wave): B[d][key] = K[key][d] -> [n][k] tile.
    v8f sacc[4] = {};
#pragma unroll
    for (int ks = 0; ks < 64; ks += 32) {
      v16bf a = ldsA(&Qs[(qrow_base + c) * LQ + ks], lane);
#pragma unroll
      for (int j = 0; j < 4; ++j) {
        v16bf bb = ldsB_nk(Ks, j * 16, ks, lane, LQ);
        sacc[j] = wmma_bf16(a, bb, sacc[j]);
      }
    }

    // Online softmax: rows live in 16-lane groups -> xor-reduce masks 1..8.
#pragma unroll
    for (int r = 0; r < 8; ++r) {
      float v = fmaxf(fmaxf(sacc[0][r], sacc[1][r]),
                      fmaxf(sacc[2][r], sacc[3][r]));
#pragma unroll
      for (int off = 1; off < 16; off <<= 1) v = fmaxf(v, __shfl_xor(v, off, 32));
      const float mnew = fmaxf(mrun[r], v);
      const float corr = __expf(mrun[r] - mnew);
      float rsum = 0.0f;
#pragma unroll
      for (int j = 0; j < 4; ++j) {
        float p = __expf(sacc[j][r] - mnew);
        sacc[j][r] = p;
        rsum += p;
      }
#pragma unroll
      for (int off = 1; off < 16; off <<= 1) rsum += __shfl_xor(rsum, off, 32);
      lrun[r] = lrun[r] * corr + rsum;
      mrun[r] = mnew;
#pragma unroll
      for (int j = 0; j < 4; ++j) o[j][r] *= corr;
      const int m_local = ((lane >> 4) << 3) + r;
#pragma unroll
      for (int j = 0; j < 4; ++j)
        Pw[m_local * LQ + j * 16 + c] = f2bf(sacc[j][r]);
    }
    __syncthreads();                            // publish P

    // O += P V : A = P (16q x 64k), B[key][d] = V[key][d] -> [k][n] tile.
#pragma unroll
    for (int ks = 0; ks < 64; ks += 32) {
      v16bf a = ldsA(&Pw[c * LQ + ks], lane);
#pragma unroll
      for (int j = 0; j < 4; ++j) {
        v16bf bb = ldsB_kn(Vs, j * 16, ks, lane, LQ);
        o[j] = wmma_bf16(a, bb, o[j]);
      }
    }
  }

  // Normalize and write ctx (bf16, [B,S,E]).
#pragma unroll
  for (int r = 0; r < 8; ++r) {
    const int m_local = ((lane >> 4) << 3) + r;
    const int s = q0 + qrow_base + m_local;
    const float inv = 1.0f / lrun[r];
    const size_t base = ((size_t)b * CS + s) * CE + head * 64;
#pragma unroll
    for (int j = 0; j < 4; ++j)
      ctx[base + j * 16 + c] = f2bf(o[j][r] * inv);
  }
}

// ---------------------------------------------------------------------------
// Kernel 3: output projection + bias.  grid = (CBS/128, CE/64), block = 256.
// ---------------------------------------------------------------------------
__global__ void __launch_bounds__(256) oproj_kernel(
    const bf16_t* __restrict__ Xc, const bf16_t* __restrict__ Wo,
    const float* __restrict__ bo, float* __restrict__ out) {
  constexpr int LDK = 40;
  __shared__ bf16_t Xs[128 * LDK];
  __shared__ bf16_t Ws[64 * LDK];

  const int tid = threadIdx.x;
  const int wid = tid >> 5, lane = tid & 31;
  const int m0 = blockIdx.x * 128;
  const int n0 = blockIdx.y * 64;

  v8f acc[4] = {};

  for (int k0 = 0; k0 < CE; k0 += 32) {
    __syncthreads();
#pragma unroll
    for (int t = 0; t < 2; ++t) {
      const int ci = tid + t * 256;
      const int r = ci >> 2, ch = ci & 3;
      cp_async16(&Xs[r * LDK + ch * 8], &Xc[(size_t)(m0 + r) * CE + k0 + ch * 8]);
    }
    {
      const int r = tid >> 2, ch = tid & 3;
      cp_async16(&Ws[r * LDK + ch * 8], &Wo[(size_t)(n0 + r) * CE + k0 + ch * 8]);
    }
    wait_async();
    __syncthreads();
    v16bf a = ldsA(&Xs[(wid * 16 + (lane & 15)) * LDK], lane);
#pragma unroll
    for (int j = 0; j < 4; ++j) {
      v16bf b = ldsB_nk(Ws, j * 16, 0, lane, LDK);
      acc[j] = wmma_bf16(a, b, acc[j]);
    }
  }

  const int c = lane & 15;
#pragma unroll
  for (int r = 0; r < 8; ++r) {
    const int m_local = ((lane >> 4) << 3) + r;
    const int gm = m0 + wid * 16 + m_local;
#pragma unroll
    for (int j = 0; j < 4; ++j) {
      const int col = n0 + j * 16 + c;
      out[(size_t)gm * CE + col] = acc[j][r] + bo[col];
    }
  }
}

// ---------------------------------------------------------------------------
extern "C" void kernel_launch(void* const* d_in, const int* in_sizes, int n_in,
                              void* d_out, int out_size, void* d_ws, size_t ws_size,
                              hipStream_t stream) {
  const float* hs   = (const float*)d_in[0];
  const float* rope = (const float*)d_in[1];
  const float* Wq   = (const float*)d_in[2];
  const float* Wk   = (const float*)d_in[3];
  const float* Wv   = (const float*)d_in[4];
  const float* Wo   = (const float*)d_in[5];
  const float* bo   = (const float*)d_in[6];
  float* out = (float*)d_out;

  // Workspace (all bf16):
  //  hsb 16MB | Wqb/Wkb/Wvb/Wob 2MB ea | Qb/Kb/Vb 16MB ea | ctx 16MB = 88MB
  char* ws = (char*)d_ws;
  const size_t hs_elems = (size_t)CBS * CE;     // 8.4M
  const size_t w_elems  = (size_t)CE * CE;      // 1.0M
  bf16_t* hsb = (bf16_t*)(ws);
  bf16_t* Wqb = hsb + hs_elems;
  bf16_t* Wkb = Wqb + w_elems;
  bf16_t* Wvb = Wkb + w_elems;
  bf16_t* Wob = Wvb + w_elems;
  bf16_t* Qb  = Wob + w_elems;
  bf16_t* Kb  = Qb + hs_elems;
  bf16_t* Vb  = Kb + hs_elems;
  bf16_t* ctx = Vb + hs_elems;

  // k0: one-time fp32 -> bf16 conversions (memory-bound, ~2us total).
  convert_bf16_kernel<<<(int)(hs_elems / 2048), 256, 0, stream>>>(hs, hsb, (int)hs_elems);
  convert_bf16_kernel<<<(int)(w_elems / 2048), 256, 0, stream>>>(Wq, Wqb, (int)w_elems);
  convert_bf16_kernel<<<(int)(w_elems / 2048), 256, 0, stream>>>(Wk, Wkb, (int)w_elems);
  convert_bf16_kernel<<<(int)(w_elems / 2048), 256, 0, stream>>>(Wv, Wvb, (int)w_elems);
  convert_bf16_kernel<<<(int)(w_elems / 2048), 256, 0, stream>>>(Wo, Wob, (int)w_elems);

  dim3 g1(CBS / 128, CH, 3);
  qkv_rope_kernel<<<g1, 256, 0, stream>>>(hsb, rope, Wqb, Wkb, Wvb, Qb, Kb, Vb);

  dim3 g2(CS / 64, CH, CB);
  attention_kernel<<<g2, 128, 0, stream>>>(Qb, Kb, Vb, ctx);

  dim3 g3(CBS / 128, CE / 64, 1);
  oproj_kernel<<<g3, 256, 0, stream>>>(ctx, Wob, bo, out);
}
